// DualBranchGCN_16595753631949
// MI455X (gfx1250) — matrix operand order repaired and checked by reading, not compiled, not run on hardware
//
#include <hip/hip_runtime.h>
#include <hip/hip_bf16.h>

// ---------------------------------------------------------------------------
// DualBranchGCN on MI455X (gfx1250): all GEMMs via v_wmma_f32_16x16x32_bf16.
// 128x128 WG tile, 8 waves x (32x64). Single fully-vectorized staging path:
// conv weights are pre-repacked, dilated-conv shifts are applied to the
// OUTPUT columns in the epilogue (bounds-checked scalar stores).
// ---------------------------------------------------------------------------

typedef __attribute__((ext_vector_type(16))) __bf16 v16bf;
typedef __attribute__((ext_vector_type(8)))  __bf16 v8bf;
typedef __attribute__((ext_vector_type(8)))  float  v8f;

namespace cfg {
constexpr int B  = 64;
constexpr int N  = 256;   // nodes (== channels for TCN)
constexpr int H  = 4;     // heads
constexpr int O  = 256;   // NOUT == NHID == NFEAT
constexpr float ALPHA   = 0.2f;
constexpr float NEG_BIG = -9.0e15f;
constexpr float INV_BN  = 0.9999950000374997f; // 1/sqrt(1+1e-5)
}

// pack two f32 -> bf16x2 dword (round-half-up + v_perm_b32 byte select)
__device__ __forceinline__ unsigned pack2bf(float lo, float hi) {
  const unsigned ul = __float_as_uint(lo) + 0x8000u;
  const unsigned uh = __float_as_uint(hi) + 0x8000u;
  return __builtin_amdgcn_perm(uh, ul, 0x07060302u);
}

union Frag16 { v8bf h[2]; v16bf v; };

// ------------------------------ generic GEMM -------------------------------
// R = A[z] (MxK, row-major) * B[z] (KxN, row-major);  N == M == 256 tiles.
// Output column shift: C[m][n + outShift] (op)= act(R[m][n] + bias), columns
// falling outside [0,256) are dropped (zero-padded dilated conv taps).
struct GP {
  const float* A; const float* Bm; float* C; const float* bias;
  int lda, ldb, ldc;
  long long aB1, aB2, bB1, bB2, cB1, cB2;
  int K, Z2, act, accum, outShift;   // act: 0 none, 1 relu, 2 elu
};

__global__ __launch_bounds__(256) void wmma_gemm_bf16(GP p) {
  __shared__ alignas(16) __bf16 As[128][40];  // [m][k], 40-elem row => 20-bank stride
  __shared__ alignas(16) __bf16 Bs[128][40];  // [n][k] (transposed during staging)

  const int tid  = threadIdx.x;
  const int lane = tid & 31;
  const int w    = tid >> 5;
  const int z1 = blockIdx.z / p.Z2, z2 = blockIdx.z % p.Z2;

  const float* Ab = p.A  + (long long)z1 * p.aB1 + (long long)z2 * p.aB2;
  const float* Bb = p.Bm + (long long)z1 * p.bB1 + (long long)z2 * p.bB2;
  float*       C  = p.C  + (long long)z1 * p.cB1 + (long long)z2 * p.cB2;

  const int mWG = blockIdx.y * 128;
  const int nWG = blockIdx.x * 128;
  const int mOff = (w >> 1) * 32;   // 4 waves along M
  const int nOff = (w & 1) * 64;    // 2 waves along N

  v8f acc[2][4];
  #pragma unroll
  for (int i = 0; i < 2; ++i)
    #pragma unroll
    for (int j = 0; j < 4; ++j) acc[i][j] = {};

  // cooperative staging indices
  const int arow  = tid >> 1;        // 0..127
  const int acolh = (tid & 1) * 16;  // 0 or 16
  const int bk4   = (tid >> 5) * 4;  // 0..28  (k group)
  const int bn4   = (tid & 31) * 4;  // 0..124 (n group)

  // loop-invariant global pointers (advance by pointer arithmetic only)
  const float* ap = Ab + (size_t)(mWG + arow) * p.lda + acolh;
  const float* bp = Bb + (size_t)bk4 * p.ldb + nWG + bn4;
  const size_t bstep = (size_t)32 * p.ldb;

  // loop-invariant LDS pointers
  const int row = lane & 15;
  const int ca  = (lane < 16) ? 0 : 2; // A chunks: {ca, ca+1}
  const int cb  = (lane < 16) ? 0 : 2; // B chunks: {cb, cb+1}
  const v8bf* a0p = (const v8bf*)(&As[mOff + row][0]);
  const v8bf* a1p = (const v8bf*)(&As[mOff + 16 + row][0]);
  const v8bf* b0p = (const v8bf*)(&Bs[nOff + row][0]);
  const v8bf* b1p = (const v8bf*)(&Bs[nOff + 16 + row][0]);
  const v8bf* b2p = (const v8bf*)(&Bs[nOff + 32 + row][0]);
  const v8bf* b3p = (const v8bf*)(&Bs[nOff + 48 + row][0]);
  uint4* asd0 = (uint4*)(&As[arow][acolh]);
  uint4* asd1 = (uint4*)(&As[arow][acolh + 8]);
  uint2* bsd0 = (uint2*)(&Bs[bn4 + 0][bk4]);
  uint2* bsd1 = (uint2*)(&Bs[bn4 + 1][bk4]);
  uint2* bsd2 = (uint2*)(&Bs[bn4 + 2][bk4]);
  uint2* bsd3 = (uint2*)(&Bs[bn4 + 3][bk4]);

  for (int kb = 0; kb < p.K; kb += 32) {
    // ---- stage A tile 128x32 -> bf16 (16 elems/thread) ----
    {
      const float4 f0 = *(const float4*)(ap + 0);
      const float4 f1 = *(const float4*)(ap + 4);
      const float4 f2 = *(const float4*)(ap + 8);
      const float4 f3 = *(const float4*)(ap + 12);
      uint4 q0, q1;
      q0.x = pack2bf(f0.x, f0.y); q0.y = pack2bf(f0.z, f0.w);
      q0.z = pack2bf(f1.x, f1.y); q0.w = pack2bf(f1.z, f1.w);
      q1.x = pack2bf(f2.x, f2.y); q1.y = pack2bf(f2.z, f2.w);
      q1.z = pack2bf(f3.x, f3.y); q1.w = pack2bf(f3.z, f3.w);
      *asd0 = q0;
      *asd1 = q1;
    }
    // ---- stage B tile 32x128, transposed -> Bs[n][k] (4k x 4n block) ----
    {
      const float4 r0 = *(const float4*)(bp);
      const float4 r1 = *(const float4*)(bp + p.ldb);
      const float4 r2 = *(const float4*)(bp + 2 * p.ldb);
      const float4 r3 = *(const float4*)(bp + 3 * p.ldb);
      uint2 c0, c1, c2, c3;
      c0.x = pack2bf(r0.x, r1.x); c0.y = pack2bf(r2.x, r3.x);
      c1.x = pack2bf(r0.y, r1.y); c1.y = pack2bf(r2.y, r3.y);
      c2.x = pack2bf(r0.z, r1.z); c2.y = pack2bf(r2.z, r3.z);
      c3.x = pack2bf(r0.w, r1.w); c3.y = pack2bf(r2.w, r3.w);
      *bsd0 = c0; *bsd1 = c1; *bsd2 = c2; *bsd3 = c3;
    }
    ap += 32;
    bp += bstep;
    __syncthreads();

    // ---- fragments + 8 WMMAs ----
    Frag16 a[2], b[4];
    a[0].h[0] = a0p[ca]; a[0].h[1] = a0p[ca + 1];
    a[1].h[0] = a1p[ca]; a[1].h[1] = a1p[ca + 1];
    b[0].h[0] = b0p[cb]; b[0].h[1] = b0p[cb + 1];
    b[1].h[0] = b1p[cb]; b[1].h[1] = b1p[cb + 1];
    b[2].h[0] = b2p[cb]; b[2].h[1] = b2p[cb + 1];
    b[3].h[0] = b3p[cb]; b[3].h[1] = b3p[cb + 1];
    #pragma unroll
    for (int i = 0; i < 2; ++i)
      #pragma unroll
      for (int j = 0; j < 4; ++j)
        acc[i][j] = __builtin_amdgcn_wmma_f32_16x16x32_bf16(
            false, a[i].v, false, b[j].v, (short)0, acc[i][j], false, false);
    __syncthreads();
  }

  // ---- epilogue (output-column shift for dilated taps) ----
  const int col    = lane & 15;
  const int rowOff = (lane < 16) ? 0 : 8;
  #pragma unroll
  for (int j = 0; j < 4; ++j) {
    const int n = nWG + nOff + j * 16 + col + p.outShift;
    if ((unsigned)n >= 256u) continue;   // dropped (zero-padding region)
    const float bn = p.bias ? p.bias[n] : 0.0f;
    #pragma unroll
    for (int i = 0; i < 2; ++i) {
      #pragma unroll
      for (int r = 0; r < 8; ++r) {
        const int m = mWG + mOff + i * 16 + rowOff + r;
        const size_t idx = (size_t)m * p.ldc + n;
        float v = acc[i][j][r] + bn;
        if (p.accum) v += C[idx];
        if (p.act == 1)      v = fmaxf(v, 0.0f);
        else if (p.act == 2) v = v > 0.0f ? v : __expf(v) - 1.0f;
        C[idx] = v;
      }
    }
  }
}

// ---------------- conv-weight repack: (O,I,k) -> k x (O,I) -----------------
__global__ __launch_bounds__(256) void repack_conv(const float* __restrict__ cw,
                                                   float* __restrict__ wt,
                                                   int k, int total) {
  const int t = blockIdx.x * blockDim.x + threadIdx.x;
  if (t >= total) return;
  const int i = t & 255;
  const int o = (t >> 8) & 255;
  const int j = t >> 16;
  wt[t] = cw[(size_t)(((o << 8) + i)) * k + j];
}

// ------------------------- GAT score dot products --------------------------
__global__ __launch_bounds__(256) void gat_scores(const float* __restrict__ Wh,
                                                  const float* __restrict__ att_a,
                                                  float* __restrict__ e1,
                                                  float* __restrict__ e2) {
  const int gw = (blockIdx.x * blockDim.x + threadIdx.x) >> 5;
  const int lane = threadIdx.x & 31;
  if (gw >= cfg::B * cfg::H * cfg::N) return;
  const int h = (gw / cfg::N) % cfg::H;
  const float* rowp = Wh + (size_t)gw * cfg::O;
  const float* a1 = att_a + (size_t)h * 2 * cfg::O;
  const float* a2 = a1 + cfg::O;
  float s1 = 0.0f, s2 = 0.0f;
  for (int o = lane; o < cfg::O; o += 32) {
    const float wv = rowp[o];
    s1 += wv * a1[o];
    s2 += wv * a2[o];
  }
  #pragma unroll
  for (int off = 16; off > 0; off >>= 1) {
    s1 += __shfl_xor(s1, off, 32);
    s2 += __shfl_xor(s2, off, 32);
  }
  if (lane == 0) { e1[gw] = s1; e2[gw] = s2; }
}

// ------------------- masked LeakyReLU + row softmax ------------------------
__global__ __launch_bounds__(256) void gat_softmax(const float* __restrict__ e1,
                                                   const float* __restrict__ e2,
                                                   const float* __restrict__ adj,
                                                   float* __restrict__ att) {
  const int gw = (blockIdx.x * blockDim.x + threadIdx.x) >> 5;
  const int lane = threadIdx.x & 31;
  if (gw >= cfg::B * cfg::H * cfg::N) return;
  const int n  = gw % cfg::N;
  const int bh = gw / cfg::N;
  const float* e2r  = e2 + (size_t)bh * cfg::N;
  const float* adjr = adj + (size_t)n * cfg::N;
  const float ev1 = e1[gw];

  float vals[8];
  float mx = -3.4e38f;
  #pragma unroll
  for (int i = 0; i < 8; ++i) {
    const int m = lane + (i << 5);
    float v = ev1 + e2r[m];
    v = (v > 0.0f) ? v : cfg::ALPHA * v;
    v = (adjr[m] > 0.0f) ? v : cfg::NEG_BIG;
    vals[i] = v;
    mx = fmaxf(mx, v);
  }
  #pragma unroll
  for (int off = 16; off > 0; off >>= 1) mx = fmaxf(mx, __shfl_xor(mx, off, 32));
  float sum = 0.0f;
  #pragma unroll
  for (int i = 0; i < 8; ++i) { vals[i] = __expf(vals[i] - mx); sum += vals[i]; }
  #pragma unroll
  for (int off = 16; off > 0; off >>= 1) sum += __shfl_xor(sum, off, 32);
  const float inv = 1.0f / sum;
  float* outr = att + (size_t)gw * cfg::N;
  #pragma unroll
  for (int i = 0; i < 8; ++i) outr[lane + (i << 5)] = vals[i] * inv;
}

// -------------------- elementwise epilogues (BN/ReLU) ----------------------
__global__ __launch_bounds__(256) void tcn_finish(const float* __restrict__ y,
                                                  const float* __restrict__ res,
                                                  const float* __restrict__ cb,
                                                  const float* __restrict__ g,
                                                  const float* __restrict__ bb,
                                                  float* __restrict__ out, int total) {
  const int i = blockIdx.x * blockDim.x + threadIdx.x;
  if (i >= total) return;
  const int o = (i >> 8) & 255;
  float v = (y[i] + cb[o]) * cfg::INV_BN;
  v = fmaxf(g[o] * v + bb[o], 0.0f);
  out[i] = v + res[i];
}

__global__ __launch_bounds__(256) void fusion_finish(const float* __restrict__ y,
                                                     const float* __restrict__ fb,
                                                     const float* __restrict__ g,
                                                     const float* __restrict__ bb,
                                                     float* __restrict__ out, int total) {
  const int i = blockIdx.x * blockDim.x + threadIdx.x;
  if (i >= total) return;
  const int o = (i >> 8) & 255;
  float v = (y[i] + fb[o]) * cfg::INV_BN;
  out[i] = fmaxf(g[o] * v + bb[o], 0.0f);
}

// ------------------------------- host side ---------------------------------
namespace {

void gemm(hipStream_t st,
          const float* A, int lda, long long aB1, long long aB2,
          const float* Bm, int ldb, long long bB1, long long bB2,
          float* C, int ldc, long long cB1, long long cB2,
          const float* bias, int K, int Z, int Z2, int act, int accum,
          int outShift = 0) {
  GP p;
  p.A = A; p.Bm = Bm; p.C = C; p.bias = bias;
  p.lda = lda; p.ldb = ldb; p.ldc = ldc;
  p.aB1 = aB1; p.aB2 = aB2; p.bB1 = bB1; p.bB2 = bB2; p.cB1 = cB1; p.cB2 = cB2;
  p.K = K; p.Z2 = Z2; p.act = act; p.accum = accum; p.outShift = outShift;
  dim3 grid(2, 2, Z);
  wmma_gemm_bf16<<<grid, dim3(256), 0, st>>>(p);
}

struct BranchParams {
  const float *gcn_w, *gcn_b, *gcn_adj, *att_w, *att_a, *fc_w, *fc_b;
  const float *conv_w, *conv_b, *bn_g, *bn_b, *res_w;
};

struct Scratch {
  float *S, *F, *Wh, *att, *hpC, *e1, *e2, *wtap;  // S reused as G, F as res, att as y
};

void run_branch(hipStream_t st, const float* x, const float* adj,
                const BranchParams& bp, int ktaps, int dil,
                const Scratch& s, float* branch_out) {
  using namespace cfg;
  const long long M2 = (long long)N * O;          // 65536
  const int Zb = B, Zbh = B * H;
  const int nRowWaves = B * H * N;                // 65536
  const int waveBlocks = (nRowWaves * 32) / 256;  // 8192
  const int total = B * N * O;
  const int ewBlocks = (total + 255) / 256;

  // 1) support = x @ gcn_w
  gemm(st, x, O, M2, 0,  bp.gcn_w, O, 0, 0,
       s.S, O, M2, 0,  nullptr, O, Zb, 1, 0, 0);
  // 2) F = relu(gcn_adj @ support + gcn_b)
  gemm(st, bp.gcn_adj, N, 0, 0,  s.S, O, M2, 0,
       s.F, O, M2, 0,  bp.gcn_b, N, Zb, 1, 1, 0);
  // 3) Wh[b,h] = F[b] @ att_w[h]
  gemm(st, s.F, O, M2, 0,  bp.att_w, O, 0, M2,
       s.Wh, O, (long long)H * M2, M2,  nullptr, O, Zbh, H, 0, 0);
  // 4) score dots
  gat_scores<<<waveBlocks, 256, 0, st>>>(s.Wh, bp.att_a, s.e1, s.e2);
  // 5) masked softmax -> att
  gat_softmax<<<waveBlocks, 256, 0, st>>>(s.e1, s.e2, adj, s.att);
  // 6) hp = elu(att @ Wh), written directly in concat layout (B,N,H*O)
  gemm(st, s.att, N, (long long)H * M2, M2,  s.Wh, O, (long long)H * M2, M2,
       s.hpC, H * O, (long long)N * H * O, O,  nullptr, N, Zbh, H, 2, 0);
  // 7) G = hpC @ fc_w + fc_b   (K = H*O = 1024), reuse S as G
  gemm(st, s.hpC, H * O, (long long)N * H * O, 0,  bp.fc_w, O, 0, 0,
       s.S, O, M2, 0,  bp.fc_b, H * O, Zb, 1, 0, 0);
  // 8) TCN. Repack conv_w -> k contiguous (O,I) matrices, then per-tap GEMMs
  //    with OUTPUT column shift (pad - j*dil); tap with shift 0 goes first.
  const int pad = (ktaps - 1) * dil / 2;
  const int wtotal = (int)(M2 * ktaps);
  repack_conv<<<(wtotal + 255) / 256, 256, 0, st>>>(bp.conv_w, s.wtap, ktaps, wtotal);
  const int jz = pad / dil;                       // tap with zero shift
  for (int t = 0; t < ktaps; ++t) {
    const int j = (t == 0) ? jz : (t <= jz ? t - 1 : t);  // jz first, others follow
    gemm(st, s.wtap + (long long)j * M2, O, 0, 0,
         s.S, O, M2, 0,
         s.att /*y*/, O, M2, 0,  nullptr, O, Zb, 1, 0, t > 0, pad - j * dil);
  }
  gemm(st, bp.res_w, O, 0, 0,  s.S, O, M2, 0,
       s.F /*res*/, O, M2, 0,  nullptr, O, Zb, 1, 0, 0);
  // 9) out = relu(bn(y + conv_b)) + res
  tcn_finish<<<ewBlocks, 256, 0, st>>>(s.att, s.F, bp.conv_b, bp.bn_g, bp.bn_b,
                                       branch_out, total);
}

BranchParams unpack_branch(void* const* d_in, int base) {
  BranchParams p;
  p.gcn_w  = (const float*)d_in[base + 0];
  p.gcn_b  = (const float*)d_in[base + 1];
  p.gcn_adj= (const float*)d_in[base + 2];
  p.att_w  = (const float*)d_in[base + 3];
  p.att_a  = (const float*)d_in[base + 4];
  p.fc_w   = (const float*)d_in[base + 5];
  p.fc_b   = (const float*)d_in[base + 6];
  p.conv_w = (const float*)d_in[base + 7];
  p.conv_b = (const float*)d_in[base + 8];
  p.bn_g   = (const float*)d_in[base + 9];
  p.bn_b   = (const float*)d_in[base + 10];
  p.res_w  = (const float*)d_in[base + 11];
  return p;
}

} // namespace

extern "C" void kernel_launch(void* const* d_in, const int* in_sizes, int n_in,
                              void* d_out, int out_size, void* d_ws, size_t ws_size,
                              hipStream_t stream) {
  using namespace cfg;
  (void)in_sizes; (void)n_in; (void)out_size; (void)ws_size;

  const float* x   = (const float*)d_in[0];
  const float* adj = (const float*)d_in[1];
  const BranchParams micro = unpack_branch(d_in, 2);
  const BranchParams macro = unpack_branch(d_in, 14);
  const float* fus_w  = (const float*)d_in[26];
  const float* fus_b  = (const float*)d_in[27];
  const float* fus_g  = (const float*)d_in[28];
  const float* fus_bb = (const float*)d_in[29];

  // workspace layout (floats)
  float* W = (float*)d_ws;
  const size_t M2  = (size_t)N * O;          // 65536
  const size_t SZ1 = (size_t)B * M2;         // 4 Mi floats
  const size_t SZ4 = (size_t)B * H * M2;     // 16 Mi floats
  Scratch s;
  size_t off = 0;
  s.S    = W + off; off += SZ1;
  s.F    = W + off; off += SZ1;
  s.Wh   = W + off; off += SZ4;
  s.att  = W + off; off += SZ4;
  s.hpC  = W + off; off += SZ4;
  s.e1   = W + off; off += (size_t)B * H * N;
  s.e2   = W + off; off += (size_t)B * H * N;
  s.wtap = W + off; off += 5 * M2;           // repacked conv weights (k<=5)
  float* mi = W + off; off += SZ1;
  float* ma = W + off; off += SZ1;
  float* yF = s.Wh;  // fusion conv accumulator (Wh free by then)

  // two branches: micro (k=3, dil=2), macro (k=5, dil=1)
  run_branch(stream, x, adj, micro, 3, 2, s, mi);
  run_branch(stream, x, adj, macro, 5, 1, s, ma);

  // fusion: y = W[:, :256] @ mi + W[:, 256:] @ ma ; out = relu(bn(y + b))
  gemm(stream, fus_w, 2 * O, 0, 0,        mi, O, (long long)M2, 0,
       yF, O, (long long)M2, 0,  nullptr, O, B, 1, 0, 0);
  gemm(stream, fus_w + O, 2 * O, 0, 0,    ma, O, (long long)M2, 0,
       yF, O, (long long)M2, 0,  nullptr, O, B, 1, 0, 1);
  const int total = B * N * O;
  fusion_finish<<<(total + 255) / 256, 256, 0, stream>>>(
      yF, fus_b, fus_g, fus_bb, (float*)d_out, total);
}
